// TransformerBlock_fc_36876589203655
// MI455X (gfx1250) — matrix operand order repaired and checked by reading, not compiled, not run on hardware
//
#include <hip/hip_runtime.h>

// ---------- types & helpers ----------
typedef __attribute__((ext_vector_type(16))) __bf16 bf16x16;
typedef __attribute__((ext_vector_type(8)))  float  f32x8;

union Frag { unsigned int u[8]; bf16x16 v; };

__device__ __forceinline__ unsigned short f2bf(float f) {
  unsigned int u = __float_as_uint(f);
  u += 0x7FFFu + ((u >> 16) & 1u);       // round-to-nearest-even
  return (unsigned short)(u >> 16);
}
__device__ __forceinline__ float bf2f(unsigned short h) {
  return __uint_as_float(((unsigned int)h) << 16);
}

__device__ __forceinline__ f32x8 wmma_bf16(bf16x16 a, bf16x16 b, f32x8 c) {
  return __builtin_amdgcn_wmma_f32_16x16x32_bf16(false, a, false, b, (short)0, c,
                                                 false, false);
}

// A fragment (16x32 bf16, row-major tile, rowDw dwords/row).
// lane: m = L&15, kh = L>>4. VGPR v: K = kh*16+2v (8 contiguous dwords).
__device__ __forceinline__ bf16x16 load_frag_A(const unsigned int* base, int m,
                                               int kh, int rowDw, int kDwOff) {
  Frag f;
  const unsigned int* p0 = base + m * rowDw + kDwOff + kh * 8;
  const unsigned int* p1 = p0 + 4;
#pragma unroll
  for (int i = 0; i < 4; i++) { f.u[i] = p0[i]; f.u[4 + i] = p1[i]; }
  return f.v;
}

// B fragment from row-major W[n_out][k_in] bf16 (exactly x@W.T layout).
// lane: n = n0 + (L&15); VGPR v: K = (L>>4)*16 + 2v  -> 8 contiguous dwords.
__device__ __forceinline__ bf16x16 load_frag_B(const unsigned int* base, int n0,
                                               int nn, int kh, int rowDw,
                                               int kDwOff) {
  Frag f;
  const unsigned int* p = base + (n0 + nn) * rowDw + kDwOff + kh * 8;
#pragma unroll
  for (int i = 0; i < 8; i++) f.u[i] = p[i];
  return f.v;
}

// Async global->LDS copy of 16B per lane (ASYNCcnt-tracked, CDNA5).
__device__ __forceinline__ void async_copy_b128(unsigned int ldsOff,
                                                unsigned int gOff,
                                                unsigned long long gBase) {
  asm volatile("global_load_async_to_lds_b128 %0, %1, %2 offset:0"
               :
               : "v"(ldsOff), "v"(gOff), "s"(gBase)
               : "memory");
}

// ---------- weight conversion ----------
__global__ void k_cvt(const float* __restrict__ s, unsigned short* __restrict__ d, int n) {
  int i = blockIdx.x * 256 + threadIdx.x;
  if (i < n) d[i] = f2bf(s[i]);
}
// Wd1 [64][60] -> bf16 [64][64] zero-padded
__global__ void k_cvt_pad(const float* __restrict__ s, unsigned short* __restrict__ d) {
  int i = blockIdx.x * 256 + threadIdx.x;
  if (i < 4096) {
    int r = i >> 6, c = i & 63;
    d[i] = (c < 60) ? f2bf(s[r * 60 + c]) : (unsigned short)0;
  }
}

// ---------- KNN: top-24 by squared distance ----------
__global__ void k_knn(const float* __restrict__ xyz, int* __restrict__ knn) {
  __shared__ float sx[256], sy[256], sz[256];
  const int b = blockIdx.y;
  const int q = blockIdx.x * 256 + threadIdx.x;
  const int base = b * 8192;
  const float qx = xyz[(base + q) * 3 + 0];
  const float qy = xyz[(base + q) * 3 + 1];
  const float qz = xyz[(base + q) * 3 + 2];
  float bd[24]; int bi[24];
#pragma unroll
  for (int i = 0; i < 24; i++) { bd[i] = 3.4e38f; bi[i] = 0; }
  float wd = 3.4e38f;  // register copy of bd[23] for the hot compare
  for (int tile = 0; tile < 8192; tile += 256) {
    int p = tile + threadIdx.x;
    sx[threadIdx.x] = xyz[(base + p) * 3 + 0];
    sy[threadIdx.x] = xyz[(base + p) * 3 + 1];
    sz[threadIdx.x] = xyz[(base + p) * 3 + 2];
    __syncthreads();
#pragma unroll 4
    for (int j = 0; j < 256; j++) {
      float dx = qx - sx[j], dy = qy - sy[j], dz = qz - sz[j];
      float d = dx * dx + dy * dy + dz * dz;
      if (d < wd) {
        int p2 = 23;
        while (p2 > 0 && bd[p2 - 1] > d) {
          bd[p2] = bd[p2 - 1]; bi[p2] = bi[p2 - 1]; --p2;
        }
        bd[p2] = d; bi[p2] = tile + j;
        wd = bd[23];
      }
    }
    __syncthreads();
  }
#pragma unroll
  for (int i = 0; i < 24; i++) knn[(base + q) * 24 + i] = base + bi[i];
}

// ---------- fc1: x = features @ W1.T + b1   (K=32 -> one WMMA per tile) ----------
__global__ void k_fc1(const float* __restrict__ feat,
                      const unsigned short* __restrict__ W1bf,
                      const float* __restrict__ b1, float* __restrict__ x) {
  __shared__ unsigned int sW[64 * 16];       // W1 bf16 [64][32] as dwords
  __shared__ float sb1[64];
  __shared__ unsigned int sA[8][16 * 16];    // per-wave 16x32 bf16
  const int tid = threadIdx.x, wave = tid >> 5, L = tid & 31;
  const int m = L & 15, kh = L >> 4, nn = m;
  const unsigned int* W1d = (const unsigned int*)W1bf;
  for (int i = tid; i < 1024; i += 256) sW[i] = W1d[i];
  if (tid < 64) sb1[tid] = b1[tid];
  const int t0 = (blockIdx.x * 8 + wave) * 16;
  {
    const float* src = feat + (t0 + m) * 32 + kh * 16;
    unsigned int* dst = &sA[wave][m * 16 + kh * 8];
#pragma unroll
    for (int i = 0; i < 8; i++) {
      unsigned int lo = f2bf(src[2 * i]), hi = f2bf(src[2 * i + 1]);
      dst[i] = lo | (hi << 16);
    }
  }
  __syncthreads();
  bf16x16 a = load_frag_A(&sA[wave][0], m, kh, 16, 0);
#pragma unroll
  for (int n0 = 0; n0 < 64; n0 += 16) {
    f32x8 c = {0, 0, 0, 0, 0, 0, 0, 0};
    bf16x16 b = load_frag_B(sW, n0, nn, kh, 16, 0);
    c = wmma_bf16(a, b, c);
    float bias = sb1[n0 + nn];
#pragma unroll
    for (int v = 0; v < 8; v++) {
      int mm = v + 8 * kh;
      x[(t0 + mm) * 64 + n0 + nn] = c[v] + bias;
    }
  }
}

// ---------- pos_enc + pre: sincos embed -> fc_delta (2x WMMA GEMMs) ----------
__global__ void k_posenc(const float* __restrict__ xyz, const int* __restrict__ knn,
                         const float* __restrict__ x,
                         const unsigned short* __restrict__ Wd1bf,
                         const float* __restrict__ bd1,
                         const unsigned short* __restrict__ Wd2bf,
                         const float* __restrict__ bd2,
                         unsigned short* __restrict__ preBf,
                         unsigned short* __restrict__ peBf) {
  __shared__ unsigned int sW1[64 * 32];      // Wd1 [64][64] bf16
  __shared__ unsigned int sW2[64 * 32];      // Wd2 [64][64] bf16
  __shared__ float sb1[64], sb2[64];
  __shared__ unsigned int sPE[8][16 * 32];   // per-wave 16x64 bf16
  __shared__ unsigned int sH[8][16 * 32];    // per-wave hidden 16x64 bf16
  __shared__ int sT[8][16], sNB[8][16];
  const int tid = threadIdx.x, wave = tid >> 5, L = tid & 31;
  const int m = L & 15, kh = L >> 4, nn = m;
  for (int i = tid; i < 2048; i += 256) {
    sW1[i] = ((const unsigned int*)Wd1bf)[i];
    sW2[i] = ((const unsigned int*)Wd2bf)[i];
  }
  if (tid < 64) { sb1[tid] = bd1[tid]; sb2[tid] = bd2[tid]; }
  const int r0 = (blockIdx.x * 8 + wave) * 16;
  const int r = r0 + m;
  const int t = r / 24;
  const int nb = knn[r];
  if (kh == 0) { sT[wave][m] = t; sNB[wave][m] = nb; }
  float g[3];
  g[0] = xyz[t * 3 + 0] - xyz[nb * 3 + 0];
  g[1] = xyz[t * 3 + 1] - xyz[nb * 3 + 1];
  g[2] = xyz[t * 3 + 2] - xyz[nb * 3 + 2];
  // each lane-half computes 30 of the 60 embedding values
  unsigned short vals[30];
#pragma unroll
  for (int i = 0; i < 30; i++) {
    int e = kh * 30 + i;
    int axis = e / 20, j = e - axis * 20;
    int jj = (j < 10) ? j : j - 10;
    float omega = __expf(-0.92103403719762f * (float)jj);  // 10000^(-jj/10)
    float ang = g[axis] * omega;
    float val = (j < 10) ? __sinf(ang) : __cosf(ang);
    vals[i] = f2bf(val);
  }
  {
    unsigned int* row = &sPE[wave][m * 32];
#pragma unroll
    for (int i = 0; i < 15; i++)
      row[kh * 15 + i] = (unsigned int)vals[2 * i] | ((unsigned int)vals[2 * i + 1] << 16);
    if (kh == 0) { row[30] = 0; row[31] = 0; }  // zero-pad cols 60..63
  }
  __syncthreads();
  // stage 1: h = relu(pe @ Wd1.T + bd1)
  bf16x16 a0 = load_frag_A(&sPE[wave][0], m, kh, 32, 0);
  bf16x16 a1 = load_frag_A(&sPE[wave][0], m, kh, 32, 16);
#pragma unroll
  for (int n0 = 0; n0 < 64; n0 += 16) {
    f32x8 c = {0, 0, 0, 0, 0, 0, 0, 0};
    c = wmma_bf16(a0, load_frag_B(sW1, n0, nn, kh, 32, 0), c);
    c = wmma_bf16(a1, load_frag_B(sW1, n0, nn, kh, 32, 16), c);
    float bias = sb1[n0 + nn];
#pragma unroll
    for (int v = 0; v < 8; v++) {
      float h = c[v] + bias; h = h > 0.0f ? h : 0.0f;
      ((unsigned short*)&sH[wave][0])[(v + 8 * kh) * 64 + n0 + nn] = f2bf(h);
    }
  }
  __syncthreads();
  // stage 2: pos_enc = h @ Wd2.T + bd2 ; emit pe_bf and pre = q - kf + pos_enc
  a0 = load_frag_A(&sH[wave][0], m, kh, 32, 0);
  a1 = load_frag_A(&sH[wave][0], m, kh, 32, 16);
#pragma unroll
  for (int n0 = 0; n0 < 64; n0 += 16) {
    f32x8 c = {0, 0, 0, 0, 0, 0, 0, 0};
    c = wmma_bf16(a0, load_frag_B(sW2, n0, nn, kh, 32, 0), c);
    c = wmma_bf16(a1, load_frag_B(sW2, n0, nn, kh, 32, 16), c);
    float bias = sb2[n0 + nn];
    int col = n0 + nn;
#pragma unroll
    for (int v = 0; v < 8; v++) {
      int mm = v + 8 * kh;
      int rr = r0 + mm;
      float pe = c[v] + bias;
      int tt = sT[wave][mm], nbb = sNB[wave][mm];
      peBf[rr * 64 + col] = f2bf(pe);
      float pre = x[tt * 64 + col] - x[nbb * 64 + col] + pe;
      preBf[rr * 64 + col] = f2bf(pre);  // rr*64 == t*1536 + k*64
    }
  }
}

// ---------- fc_atten: logits = pre @ Wa.T + ba  (M=32768, N=1536, K=1536) ----------
// Block: 256 threads (8 waves). Tile: M=64 (4 A-frags/wave), N=512 (4 col-tiles/wave).
// A tiles (64x32 bf16 = 4KB) double-buffered in LDS via global_load_async_to_lds_b128.
__global__ void k_atten(const unsigned short* __restrict__ preBf,
                        const unsigned short* __restrict__ WaBf,
                        const float* __restrict__ ba, float* __restrict__ logits) {
  __shared__ __align__(16) unsigned int sA[2][64 * 16];  // 2 x (64 rows x 32 K bf16)
  const int tid = threadIdx.x, wave = tid >> 5, L = tid & 31;
  const int m = L & 15, kh = L >> 4, nn = m;
  const int m0 = blockIdx.y * 64;
  const int cbase = blockIdx.x * 512 + wave * 64;
  const unsigned int* WaD = (const unsigned int*)WaBf;
  const unsigned long long preBase = (unsigned long long)preBf;

  // per-thread A-copy mapping: 16B each, rows of 64-byte K-slices
  const int arow = tid >> 2;           // 0..63
  const int aseg = tid & 3;            // 0..3 (16B segment within 64B row-slice)
  const unsigned int ldsOff0 =
      (unsigned int)(unsigned long long)(&sA[0][0]) + (unsigned int)tid * 16u;
  const unsigned int ldsOff1 =
      (unsigned int)(unsigned long long)(&sA[1][0]) + (unsigned int)tid * 16u;

  f32x8 acc[16];
  {
    f32x8 z = {0, 0, 0, 0, 0, 0, 0, 0};
#pragma unroll
    for (int i = 0; i < 16; i++) acc[i] = z;
  }

  // prologue: async-load tile 0 into buffer 0
  async_copy_b128(ldsOff0, (unsigned int)((m0 + arow) * 3072 + aseg * 16), preBase);

  for (int ks = 0; ks < 48; ks++) {
    const int buf = ks & 1;
    if (ks + 1 < 48) {
      // issue next tile into the other buffer, then wait for current tile only
      async_copy_b128(buf ? ldsOff0 : ldsOff1,
                      (unsigned int)((m0 + arow) * 3072 + (ks + 1) * 32 + aseg * 16),
                      preBase);
      asm volatile("s_wait_asynccnt 0x1" ::: "memory");
    } else {
      asm volatile("s_wait_asynccnt 0x0" ::: "memory");
    }
    __syncthreads();  // whole A tile visible to all waves

    bf16x16 afr[4];
#pragma unroll
    for (int mt = 0; mt < 4; mt++)
      afr[mt] = load_frag_A(&sA[buf][mt * 256], m, kh, 16, 0);
#pragma unroll
    for (int tt = 0; tt < 4; tt++) {
      bf16x16 b = load_frag_B(WaD, cbase + tt * 16, nn, kh, 768, ks * 16);
#pragma unroll
      for (int mt = 0; mt < 4; mt++)
        acc[mt * 4 + tt] = wmma_bf16(afr[mt], b, acc[mt * 4 + tt]);
    }
    __syncthreads();  // all waves done with buf before it is overwritten
  }

#pragma unroll
  for (int mt = 0; mt < 4; mt++) {
#pragma unroll
    for (int tt = 0; tt < 4; tt++) {
      int col = cbase + tt * 16 + nn;
      float bav = ba[col];
#pragma unroll
      for (int v = 0; v < 8; v++)
        logits[(m0 + mt * 16 + v + 8 * kh) * 1536 + col] = acc[mt * 4 + tt][v] + bav;
    }
  }
}

// ---------- epilogue: softmax over K, weighted sum, fc2 + residual ----------
__global__ void k_epilogue(const float* __restrict__ x, const int* __restrict__ knn,
                           const unsigned short* __restrict__ peBf,
                           const float* __restrict__ W2, const float* __restrict__ b2,
                           float* __restrict__ outRes, float* __restrict__ attn) {
  const int t = blockIdx.x;
  const int f = threadIdx.x;  // 64 threads
  __shared__ float sres[64];
  float l[24];
  float mx = -3.4e38f;
  const float scale = 0.125f;  // 1/sqrt(64)
#pragma unroll
  for (int k = 0; k < 24; k++) {
    l[k] = attn[t * 1536 + k * 64 + f] * scale;
    mx = fmaxf(mx, l[k]);
  }
  float s = 0.0f;
#pragma unroll
  for (int k = 0; k < 24; k++) { l[k] = __expf(l[k] - mx); s += l[k]; }
  float inv = 1.0f / s;
  float acc = 0.0f;
#pragma unroll
  for (int k = 0; k < 24; k++) {
    float p = l[k] * inv;
    attn[t * 1536 + k * 64 + f] = p;  // softmaxed attn written in place
    int nb = knn[t * 24 + k];
    float vpe = x[nb * 64 + f] + bf2f(peBf[(t * 24 + k) * 64 + f]);
    acc += p * vpe;
  }
  sres[f] = acc;
  __syncthreads();
  float r = b2[f] + x[t * 64 + f];
#pragma unroll 8
  for (int c = 0; c < 64; c++) r += sres[c] * W2[f * 64 + c];
  outRes[t * 64 + f] = r;
}

// ---------- launch ----------
extern "C" void kernel_launch(void* const* d_in, const int* in_sizes, int n_in,
                              void* d_out, int out_size, void* d_ws, size_t ws_size,
                              hipStream_t stream) {
  const float* features = (const float*)d_in[0];
  const float* xyz      = (const float*)d_in[1];
  const float* W1  = (const float*)d_in[2];
  const float* b1  = (const float*)d_in[3];
  const float* W2  = (const float*)d_in[4];
  const float* b2  = (const float*)d_in[5];
  const float* Wd1 = (const float*)d_in[6];
  const float* bd1 = (const float*)d_in[7];
  const float* Wd2 = (const float*)d_in[8];
  const float* bd2 = (const float*)d_in[9];
  const float* Wa  = (const float*)d_in[10];
  const float* ba  = (const float*)d_in[11];

  char* ws = (char*)d_ws;
  int*            knn   = (int*)(ws);                             //  3,145,728 B
  float*          x     = (float*)(ws + 3145728);                 //  8,388,608 B
  unsigned short* WaBf  = (unsigned short*)(ws + 11534336);       //  4,718,592 B
  unsigned short* W1Bf  = (unsigned short*)(ws + 16252928);       //      4,096 B
  unsigned short* Wd1Bf = (unsigned short*)(ws + 16257024);       //      8,192 B
  unsigned short* Wd2Bf = (unsigned short*)(ws + 16265216);       //      8,192 B
  unsigned short* preBf = (unsigned short*)(ws + 16273408);       // 100,663,296 B
  unsigned short* peBf  = (unsigned short*)(ws + 116936704);      // 100,663,296 B

  float* res  = (float*)d_out;            // [4,8192,64]
  float* attn = (float*)d_out + 2097152;  // [4,8192,24,64] (logits -> softmax in place)

  k_cvt<<<(2359296 + 255) / 256, 256, 0, stream>>>(Wa, WaBf, 2359296);
  k_cvt<<<8,  256, 0, stream>>>(W1,  W1Bf,  2048);
  k_cvt<<<16, 256, 0, stream>>>(Wd2, Wd2Bf, 4096);
  k_cvt_pad<<<16, 256, 0, stream>>>(Wd1, Wd1Bf);

  k_knn<<<dim3(32, 4), 256, 0, stream>>>(xyz, knn);
  k_fc1<<<256, 256, 0, stream>>>(features, W1Bf, b1, x);
  k_posenc<<<6144, 256, 0, stream>>>(xyz, knn, x, Wd1Bf, bd1, Wd2Bf, bd2, preBf, peBf);
  k_atten<<<dim3(3, 512), 256, 0, stream>>>(preBf, WaBf, ba, attn);
  k_epilogue<<<32768, 64, 0, stream>>>(x, knn, peBf, W2, b2, res, attn);
}
